// GTModel_12180527251761
// MI455X (gfx1250) — matrix operand order repaired and checked by reading, not compiled, not run on hardware
//
#include <hip/hip_runtime.h>
#include <hip/hip_bf16.h>

typedef _Float16 half_t;
typedef __attribute__((ext_vector_type(16))) _Float16 v16h;
typedef __attribute__((ext_vector_type(8)))  _Float16 v8h;
typedef __attribute__((ext_vector_type(8)))  float    v8f;

union V16H { v16h h; v8h p[2]; };

// ---------------------------------------------------------------------------
// f16 WMMA GEMM:  C[M,Nc] = (accum ? C : 0) + A[M,K] @ B^T,  B stored [Nc,K]
// row-major. One wave computes a 16x32 tile (two 16x16 accumulators sharing
// the A operand); K in steps of 32 via v_wmma_f32_16x16x32_f16.
// Requires: M % 16 == 0, Nc % 32 == 0, K % 32 == 0, 16-byte aligned ptrs,
// lda/ldb/ldc multiples of 8.
// ---------------------------------------------------------------------------
__global__ void __launch_bounds__(128)
k_gemm(const half_t* __restrict__ A, int lda,
       const half_t* __restrict__ B, int ldb,
       float* __restrict__ C, int ldc,
       int M, int Nc, int K, int accum)
{
    const int wave = threadIdx.x >> 5;
    const int lane = threadIdx.x & 31;
    const int hf   = lane >> 4;     // 0: lanes 0-15, 1: lanes 16-31
    const int lr   = lane & 15;
    const int m0   = blockIdx.y * 64 + wave * 16;
    const int n0   = blockIdx.x * 32;
    if (m0 >= M) return;

    v8f acc0, acc1;
    if (accum) {
#pragma unroll
        for (int r = 0; r < 8; ++r) {
            acc0[r] = C[(size_t)(m0 + r + 8 * hf) * ldc + n0 + lr];
            acc1[r] = C[(size_t)(m0 + r + 8 * hf) * ldc + n0 + 16 + lr];
        }
    } else {
#pragma unroll
        for (int r = 0; r < 8; ++r) { acc0[r] = 0.0f; acc1[r] = 0.0f; }
    }

    const half_t* arow  = A + (size_t)(m0 + lr) * lda;
    const half_t* brow0 = B + (size_t)(n0 + lr) * ldb;
    const half_t* brow1 = B + (size_t)(n0 + 16 + lr) * ldb;

    for (int k = 0; k < K; k += 32) {
        V16H a, b0, b1;
        // A 16x32 operand: lane<16 holds K {k..k+15},
        // lane>=16 holds K {k+16..k+31}  (contiguous, matches B)
        a.p[0] = *(const v8h*)(arow + k + 16 * hf);
        a.p[1] = *(const v8h*)(arow + k + 16 * hf + 8);
        // B operand col n: K values k+16*hf .. +15 contiguous in the W row
        b0.p[0] = *(const v8h*)(brow0 + k + 16 * hf);
        b0.p[1] = *(const v8h*)(brow0 + k + 16 * hf + 8);
        b1.p[0] = *(const v8h*)(brow1 + k + 16 * hf);
        b1.p[1] = *(const v8h*)(brow1 + k + 16 * hf + 8);

        acc0 = __builtin_amdgcn_wmma_f32_16x16x32_f16(
                   false, a.h, false, b0.h, (short)0, acc0, false, false);
        acc1 = __builtin_amdgcn_wmma_f32_16x16x32_f16(
                   false, a.h, false, b1.h, (short)0, acc1, false, false);
    }

#pragma unroll
    for (int r = 0; r < 8; ++r) {
        C[(size_t)(m0 + r + 8 * hf) * ldc + n0 + lr]      = acc0[r];
        C[(size_t)(m0 + r + 8 * hf) * ldc + n0 + 16 + lr] = acc1[r];
    }
}

// ---------------------------------------------------------------------------
// Elementwise / reduction helper kernels (f32 VALU)
// ---------------------------------------------------------------------------
__global__ void k_fill(float* p, float v, size_t n) {
    size_t i = (size_t)blockIdx.x * blockDim.x + threadIdx.x;
    if (i < n) p[i] = v;
}

__global__ void k_f2h(const float* __restrict__ s, half_t* __restrict__ d, size_t n) {
    size_t i = (size_t)blockIdx.x * blockDim.x + threadIdx.x;
    if (i < n) d[i] = (half_t)s[i];
}

__global__ void k_count(const long long* __restrict__ dst, float* cnt, int E) {
    int i = blockIdx.x * blockDim.x + threadIdx.x;
    if (i < E) atomicAdd(&cnt[(int)dst[i]], 1.0f);
}

__global__ void k_scatter(const float* __restrict__ h,
                          const long long* __restrict__ src,
                          const long long* __restrict__ dst,
                          float* sums, int E, int D) {
    size_t i = (size_t)blockIdx.x * blockDim.x + threadIdx.x;
    if (i >= (size_t)E * D) return;
    int e = (int)(i / D), d = (int)(i % D);
    atomicAdd(&sums[(size_t)dst[e] * D + d], h[(size_t)src[e] * D + d]);
}

__global__ void k_mean_h(const float* __restrict__ sums, const float* __restrict__ cnt,
                         half_t* __restrict__ m16, int n, int D) {
    size_t i = (size_t)blockIdx.x * blockDim.x + threadIdx.x;
    if (i >= (size_t)n * D) return;
    m16[i] = (half_t)(sums[i] / fmaxf(cnt[i / D], 1.0f));
}

// act: 0=none, 1=leaky(0.2)
__global__ void k_bias_act(const float* __restrict__ pre, const float* __restrict__ bias,
                           const float* __restrict__ res,
                           float* __restrict__ outf, half_t* __restrict__ outh,
                           size_t total, int D, int act) {
    size_t i = (size_t)blockIdx.x * blockDim.x + threadIdx.x;
    if (i >= total) return;
    int col = (int)(i % D);
    float v = pre[i];
    if (bias) v += bias[col];
    if (act == 1) v = v > 0.0f ? v : 0.2f * v;
    if (res) v += res[i];
    if (outf) outf[i] = v;
    if (outh) outh[i] = (half_t)v;
}

// Transpose the V section of qkv (with in_proj bias) into Vt [HID rows, N cols]:
// Vt[(hd*64+d)*N + node] = qkvpre[node*768 + 512 + hd*64 + d] + bias[512+...]
__global__ void k_vt(const float* __restrict__ qkvpre, const float* __restrict__ bias,
                     half_t* __restrict__ vt, int n) {
    size_t i = (size_t)blockIdx.x * blockDim.x + threadIdx.x;
    if (i >= (size_t)n * 256) return;
    int node = (int)(i / 256), c = (int)(i % 256);
    float v = qkvpre[(size_t)node * 768 + 512 + c] + bias[512 + c];
    vt[(size_t)c * n + node] = (half_t)v;
}

// One 256-thread block per row: softmax of S row (scale applied inside exp),
// write f16 probabilities to P and accumulate 0.25*p into attn-mean output.
__global__ void __launch_bounds__(256)
k_softmax(const float* __restrict__ S, half_t* __restrict__ P,
          float* __restrict__ attnOut, int n, float scale, int head) {
    const int row = blockIdx.x;
    const float* srow = S + (size_t)row * n;
    const int t = threadIdx.x, w = t >> 5, l = t & 31;
    __shared__ float sm[8];

    float mx = -3.402823e38f;
    for (int j = t; j < n; j += 256) mx = fmaxf(mx, srow[j]);
    for (int o = 16; o > 0; o >>= 1) mx = fmaxf(mx, __shfl_xor(mx, o, 32));
    if (l == 0) sm[w] = mx;
    __syncthreads();
    mx = sm[0];
#pragma unroll
    for (int i = 1; i < 8; ++i) mx = fmaxf(mx, sm[i]);
    __syncthreads();

    float sum = 0.0f;
    for (int j = t; j < n; j += 256) sum += expf((srow[j] - mx) * scale);
    for (int o = 16; o > 0; o >>= 1) sum += __shfl_xor(sum, o, 32);
    if (l == 0) sm[w] = sum;
    __syncthreads();
    sum = 0.0f;
#pragma unroll
    for (int i = 0; i < 8; ++i) sum += sm[i];
    const float inv = 1.0f / sum;

    float*  orow = attnOut + (size_t)row * n;
    half_t* prow = P + (size_t)row * n;
    for (int j = t; j < n; j += 256) {
        float pv = expf((srow[j] - mx) * scale) * inv;
        prow[j] = (half_t)pv;
        float c = 0.25f * pv;
        if (head == 0) orow[j] = c; else orow[j] += c;
    }
}

// LayerNorm over D==blockDim (256), optional linear bias before LN.
// act: 0=sigmoid, 1=tanh, 2=none
__global__ void __launch_bounds__(256)
k_ln_act(const float* __restrict__ pre, const float* __restrict__ linb,
         const float* __restrict__ g, const float* __restrict__ b,
         float* __restrict__ outf, half_t* __restrict__ outh, int D, int act) {
    const int row = blockIdx.x, t = threadIdx.x;
    float v = pre[(size_t)row * D + t];
    if (linb) v += linb[t];
    __shared__ float s[256];
    s[t] = v; __syncthreads();
    for (int o = 128; o > 0; o >>= 1) { if (t < o) s[t] += s[t + o]; __syncthreads(); }
    const float mu = s[0] / D;
    __syncthreads();
    const float d = v - mu;
    s[t] = d * d; __syncthreads();
    for (int o = 128; o > 0; o >>= 1) { if (t < o) s[t] += s[t + o]; __syncthreads(); }
    const float var = s[0] / D;
    float y = d * rsqrtf(var + 1e-5f) * g[t] + b[t];
    if (act == 0) y = 1.0f / (1.0f + expf(-y));
    else if (act == 1) y = tanhf(y);
    if (outf) outf[(size_t)row * D + t] = y;
    if (outh) outh[(size_t)row * D + t] = (half_t)y;
}

__global__ void k_gate(const float* __restrict__ f, const float* __restrict__ h,
                       const float* __restrict__ ig, const float* __restrict__ ct,
                       float* __restrict__ t, size_t n) {
    size_t i = (size_t)blockIdx.x * blockDim.x + threadIdx.x;
    if (i < n) t[i] = f[i] * h[i] + ig[i] * ct[i];
}

__global__ void k_concat(const float* __restrict__ h, const float* __restrict__ att,
                         half_t* __restrict__ comb, size_t total) {
    size_t i = (size_t)blockIdx.x * blockDim.x + threadIdx.x;
    if (i >= total) return;
    size_t row = i / 512; int j = (int)(i % 512);
    float v = (j < 256) ? h[row * 256 + j] : att[row * 256 + (j - 256)];
    comb[i] = (half_t)v;
}

// leaky(pre + b1) * rsqrt(1+eps) * bn_g + bn_b
__global__ void k_cls(const float* __restrict__ pre, const float* __restrict__ b1,
                      const float* __restrict__ g, const float* __restrict__ b,
                      float* __restrict__ out, size_t total, int D) {
    size_t i = (size_t)blockIdx.x * blockDim.x + threadIdx.x;
    if (i >= total) return;
    int col = (int)(i % D);
    float v = pre[i] + b1[col];
    v = v > 0.0f ? v : 0.2f * v;
    out[i] = v * rsqrtf(1.0f + 1e-5f) * g[col] + b[col];
}

__global__ void k_logits(const float* __restrict__ c2, const float* __restrict__ w2,
                         const float* __restrict__ b2, float* __restrict__ out, int n) {
    int row = blockIdx.x * blockDim.x + threadIdx.x;
    if (row >= n) return;
#pragma unroll
    for (int j = 0; j < 3; ++j) {
        float acc = b2[j];
        for (int k = 0; k < 128; ++k) acc += c2[(size_t)row * 128 + k] * w2[j * 128 + k];
        out[(size_t)row * 3 + j] = acc;
    }
}

// ---------------------------------------------------------------------------
extern "C" void kernel_launch(void* const* d_in, const int* in_sizes, int n_in,
                              void* d_out, int out_size, void* d_ws, size_t ws_size,
                              hipStream_t stream) {
    const int IN = 128, HID = 256, DH = 64, NCLS = 3;
    const int N = in_sizes[0] / IN;
    const int E = in_sizes[1] / 2;

    const float*     x   = (const float*)d_in[0];
    const long long* ei  = (const long long*)d_in[1];
    const long long* src = ei;
    const long long* dst = ei + E;

    const float* sage_wl[2] = {(const float*)d_in[2], (const float*)d_in[5]};
    const float* sage_wr[2] = {(const float*)d_in[3], (const float*)d_in[6]};
    const float* sage_b[2]  = {(const float*)d_in[4], (const float*)d_in[7]};
    const float* in_proj_w  = (const float*)d_in[8];
    const float* in_proj_b  = (const float*)d_in[9];
    const float* out_proj_w = (const float*)d_in[10];
    const float* out_proj_b = (const float*)d_in[11];
    const float* gw[3]  = {(const float*)d_in[12], (const float*)d_in[16], (const float*)d_in[20]};
    const float* gb[3]  = {(const float*)d_in[13], (const float*)d_in[17], (const float*)d_in[21]};
    const float* gg[3]  = {(const float*)d_in[14], (const float*)d_in[18], (const float*)d_in[22]};
    const float* gbb[3] = {(const float*)d_in[15], (const float*)d_in[19], (const float*)d_in[23]};
    const float* ln_g = (const float*)d_in[24];
    const float* ln_b = (const float*)d_in[25];
    const float* emb_w1 = (const float*)d_in[26]; const float* emb_b1 = (const float*)d_in[27];
    const float* emb_w2 = (const float*)d_in[28]; const float* emb_b2 = (const float*)d_in[29];
    const float* cls_w1 = (const float*)d_in[30]; const float* cls_b1 = (const float*)d_in[31];
    const float* bn_g   = (const float*)d_in[32]; const float* bn_b   = (const float*)d_in[33];
    const float* cls_w2 = (const float*)d_in[34]; const float* cls_b2 = (const float*)d_in[35];

    float* out        = (float*)d_out;
    float* out_emb    = out;
    float* out_logits = out + (size_t)N * (HID / 2);
    float* out_attn   = out_logits + (size_t)N * NCLS;

    char* wp = (char*)d_ws;
    auto alloc = [&](size_t bytes) -> void* {
        void* r = (void*)wp; wp += (bytes + 255) & ~(size_t)255; return r;
    };

    // f16 weight mirrors
    half_t* w16_s0l = (half_t*)alloc((size_t)HID * IN * 2);
    half_t* w16_s0r = (half_t*)alloc((size_t)HID * IN * 2);
    half_t* w16_s1l = (half_t*)alloc((size_t)HID * HID * 2);
    half_t* w16_s1r = (half_t*)alloc((size_t)HID * HID * 2);
    half_t* w16_inp = (half_t*)alloc((size_t)3 * HID * HID * 2);
    half_t* w16_out = (half_t*)alloc((size_t)HID * HID * 2);
    half_t* w16_g[3];
    for (int g = 0; g < 3; ++g) w16_g[g] = (half_t*)alloc((size_t)HID * 2 * HID * 2);
    half_t* w16_e1 = (half_t*)alloc((size_t)HID * HID * 2);
    half_t* w16_e2 = (half_t*)alloc((size_t)(HID / 2) * HID * 2);
    half_t* w16_c1 = (half_t*)alloc((size_t)(HID / 2) * (HID / 2) * 2);

    // activations
    float*  cnt    = (float*)alloc((size_t)N * 4);
    float*  sums   = (float*)alloc((size_t)N * HID * 4);
    half_t* mean16 = (half_t*)alloc((size_t)N * HID * 2);
    half_t* hin16  = (half_t*)alloc((size_t)N * HID * 2);
    float*  hpre   = (float*)alloc((size_t)N * HID * 4);
    float*  h      = (float*)alloc((size_t)N * HID * 4);
    float*  qkvpre = (float*)alloc((size_t)N * 3 * HID * 4);
    half_t* qkv16  = (half_t*)alloc((size_t)N * 3 * HID * 2);
    half_t* vt16   = (half_t*)alloc((size_t)HID * N * 2);     // V^T, [HID rows, N cols]
    float*  S      = (float*)alloc((size_t)N * N * 4);
    half_t* P16    = (half_t*)alloc((size_t)N * N * 2);
    float*  o      = (float*)alloc((size_t)N * HID * 4);
    half_t* o16    = (half_t*)alloc((size_t)N * HID * 2);
    float*  apre   = (float*)alloc((size_t)N * HID * 4);
    float*  attd   = (float*)alloc((size_t)N * HID * 4);
    half_t* comb16 = (half_t*)alloc((size_t)N * 2 * HID * 2);
    float*  gpre   = (float*)alloc((size_t)N * HID * 4);
    float*  gout[3];
    for (int g = 0; g < 3; ++g) gout[g] = (float*)alloc((size_t)N * HID * 4);
    float*  tcell  = (float*)alloc((size_t)N * HID * 4);
    float*  cell   = (float*)alloc((size_t)N * HID * 4);
    half_t* cell16 = (half_t*)alloc((size_t)N * HID * 2);
    float*  epre   = (float*)alloc((size_t)N * HID * 4);
    half_t* e16    = (half_t*)alloc((size_t)N * HID * 2);
    float*  embpre = (float*)alloc((size_t)N * (HID / 2) * 4);
    half_t* emb16  = (half_t*)alloc((size_t)N * (HID / 2) * 2);
    float*  cpre   = (float*)alloc((size_t)N * (HID / 2) * 4);
    float*  c2     = (float*)alloc((size_t)N * (HID / 2) * 4);

    auto g1 = [](size_t n) { return dim3((unsigned)((n + 255) / 256)); };
    auto f2h = [&](const float* s, half_t* d, size_t n) {
        k_f2h<<<g1(n), 256, 0, stream>>>(s, d, n);
    };
    auto gemm = [&](const half_t* A, int lda, const half_t* B, int ldb,
                    float* C, int ldc, int M, int Nc, int K, bool accum) {
        dim3 grid(Nc / 32, (M + 63) / 64), blk(128);
        k_gemm<<<grid, blk, 0, stream>>>(A, lda, B, ldb, C, ldc, M, Nc, K, (int)accum);
    };

    // weight conversions
    f2h(sage_wl[0], w16_s0l, (size_t)HID * IN);
    f2h(sage_wr[0], w16_s0r, (size_t)HID * IN);
    f2h(sage_wl[1], w16_s1l, (size_t)HID * HID);
    f2h(sage_wr[1], w16_s1r, (size_t)HID * HID);
    f2h(in_proj_w,  w16_inp, (size_t)3 * HID * HID);
    f2h(out_proj_w, w16_out, (size_t)HID * HID);
    for (int g = 0; g < 3; ++g) f2h(gw[g], w16_g[g], (size_t)HID * 2 * HID);
    f2h(emb_w1, w16_e1, (size_t)HID * HID);
    f2h(emb_w2, w16_e2, (size_t)(HID / 2) * HID);
    f2h(cls_w1, w16_c1, (size_t)(HID / 2) * (HID / 2));

    // degrees
    k_fill<<<g1(N), 256, 0, stream>>>(cnt, 0.0f, (size_t)N);
    k_count<<<(E + 255) / 256, 256, 0, stream>>>(dst, cnt, E);

    // ---- SAGE layer 0 (din=128) ----
    k_fill<<<g1((size_t)N * IN), 256, 0, stream>>>(sums, 0.0f, (size_t)N * IN);
    k_scatter<<<g1((size_t)E * IN), 256, 0, stream>>>(x, src, dst, sums, E, IN);
    k_mean_h<<<g1((size_t)N * IN), 256, 0, stream>>>(sums, cnt, mean16, N, IN);
    f2h(x, hin16, (size_t)N * IN);
    gemm(mean16, IN, w16_s0l, IN, hpre, HID, N, HID, IN, false);
    gemm(hin16,  IN, w16_s0r, IN, hpre, HID, N, HID, IN, true);
    k_bias_act<<<g1((size_t)N * HID), 256, 0, stream>>>(
        hpre, sage_b[0], nullptr, h, hin16, (size_t)N * HID, HID, 1);

    // ---- SAGE layer 1 (din=256, residual) ----
    k_fill<<<g1((size_t)N * HID), 256, 0, stream>>>(sums, 0.0f, (size_t)N * HID);
    k_scatter<<<g1((size_t)E * HID), 256, 0, stream>>>(h, src, dst, sums, E, HID);
    k_mean_h<<<g1((size_t)N * HID), 256, 0, stream>>>(sums, cnt, mean16, N, HID);
    gemm(mean16, HID, w16_s1l, HID, hpre, HID, N, HID, HID, false);
    gemm(hin16,  HID, w16_s1r, HID, hpre, HID, N, HID, HID, true);
    k_bias_act<<<g1((size_t)N * HID), 256, 0, stream>>>(
        hpre, sage_b[1], h, h, hin16, (size_t)N * HID, HID, 1);

    // ---- attention ----
    gemm(hin16, HID, w16_inp, HID, qkvpre, 3 * HID, N, 3 * HID, HID, false);
    k_bias_act<<<g1((size_t)N * 3 * HID), 256, 0, stream>>>(
        qkvpre, in_proj_b, nullptr, nullptr, qkv16, (size_t)N * 3 * HID, 3 * HID, 0);
    k_vt<<<g1((size_t)N * HID), 256, 0, stream>>>(qkvpre, in_proj_b, vt16, N);

    const float scale = 0.125f;  // 1/sqrt(64)
    for (int hd = 0; hd < 4; ++hd) {
        gemm(qkv16 + hd * DH, 3 * HID,             // Q_h [N,64]
             qkv16 + HID + hd * DH, 3 * HID,       // K_h [N,64]  (S = Q K^T)
             S, N, N, N, DH, false);
        k_softmax<<<N, 256, 0, stream>>>(S, P16, out_attn, N, scale, hd);
        gemm(P16, N,                               // P [N,N]
             vt16 + (size_t)hd * DH * N, N,        // V_h^T [64,N]  (O = P (V^T)^T)
             o + hd * DH, HID, N, DH, N, false);
    }
    f2h(o, o16, (size_t)N * HID);
    gemm(o16, HID, w16_out, HID, apre, HID, N, HID, HID, false);
    k_bias_act<<<g1((size_t)N * HID), 256, 0, stream>>>(
        apre, out_proj_b, nullptr, attd, nullptr, (size_t)N * HID, HID, 0);

    // ---- gated fusion ----
    k_concat<<<g1((size_t)N * 2 * HID), 256, 0, stream>>>(h, attd, comb16, (size_t)N * 2 * HID);
    for (int g = 0; g < 3; ++g) {
        gemm(comb16, 2 * HID, w16_g[g], 2 * HID, gpre, HID, N, HID, 2 * HID, false);
        k_ln_act<<<N, HID, 0, stream>>>(gpre, gb[g], gg[g], gbb[g],
                                        gout[g], nullptr, HID, (g == 2) ? 1 : 0);
    }
    k_gate<<<g1((size_t)N * HID), 256, 0, stream>>>(gout[0], h, gout[1], gout[2],
                                                    tcell, (size_t)N * HID);
    k_ln_act<<<N, HID, 0, stream>>>(tcell, nullptr, ln_g, ln_b, cell, cell16, HID, 2);

    // ---- embedding head ----
    gemm(cell16, HID, w16_e1, HID, epre, HID, N, HID, HID, false);
    k_bias_act<<<g1((size_t)N * HID), 256, 0, stream>>>(
        epre, emb_b1, nullptr, nullptr, e16, (size_t)N * HID, HID, 1);
    gemm(e16, HID, w16_e2, HID, embpre, HID / 2, N, HID / 2, HID, false);
    k_bias_act<<<g1((size_t)N * (HID / 2)), 256, 0, stream>>>(
        embpre, emb_b2, nullptr, out_emb, emb16, (size_t)N * (HID / 2), HID / 2, 0);

    // ---- classifier ----
    gemm(emb16, HID / 2, w16_c1, HID / 2, cpre, HID / 2, N, HID / 2, HID / 2, false);
    k_cls<<<g1((size_t)N * (HID / 2)), 256, 0, stream>>>(
        cpre, cls_b1, bn_g, bn_b, c2, (size_t)N * (HID / 2), HID / 2);
    k_logits<<<(N + 255) / 256, 256, 0, stream>>>(c2, cls_w2, cls_b2, out_logits, N);
}
